// ProtoMixer_82935818486345
// MI455X (gfx1250) — matrix operand order repaired and checked by reading, not compiled, not run on hardware
//
#include <hip/hip_runtime.h>
#include <hip/hip_bf16.h>

typedef __attribute__((ext_vector_type(16))) _Float16     v16h;
typedef __attribute__((ext_vector_type(8)))  float        v8f;
typedef __attribute__((ext_vector_type(4)))  unsigned int v4u_;
typedef __attribute__((ext_vector_type(8)))  int          v8i_;
typedef __attribute__((ext_vector_type(4)))  int          v4i_;

#define B_    256
#define M_    64
#define DSLOT 126
#define D_    128
#define C_    62
#define KP_   32
#define N_    (C_*KP_)      // 1984 prototype rows
#define NT_   (N_/16)       // 124 proto tiles
#define NCH_  (NT_/2)       // 62 chunks of 32 rows (== one class each)
#define BETA_ 4.0f
#define TOPP_ 0.9f

// ---------------- kernel 1: centers -> f16 + c2 ----------------
__global__ __launch_bounds__(128) void prep_centers_kernel(
    const float* __restrict__ centers, _Float16* __restrict__ ch, float* __restrict__ c2)
{
    int n = blockIdx.x, t = threadIdx.x;
    float v = centers[(size_t)n * D_ + t];
    ch[(size_t)n * D_ + t] = (_Float16)v;
    __shared__ float red[128];
    red[t] = v * v;
    __syncthreads();
    for (int off = 64; off; off >>= 1) {
        if (t < off) red[t] += red[t + off];
        __syncthreads();
    }
    if (t == 0) c2[n] = red[0];
}

// ---------------- kernel 2: softmax over psi (per class, one wave32) ----------------
__global__ __launch_bounds__(32) void softmax_psi_kernel(
    const float* __restrict__ psi, float* __restrict__ w)
{
    int c = blockIdx.x, l = threadIdx.x;
    float v = psi[c * KP_ + l];
    float m = v;
    for (int off = 16; off; off >>= 1) m = fmaxf(m, __shfl_xor(m, off, 32));
    float e = __expf(v - m);
    float s = e;
    for (int off = 16; off; off >>= 1) s += __shfl_xor(s, off, 32);
    w[c * KP_ + l] = e / s;
}

// ---- TDM: 1-D DMA of `bytes` (multiple of 4, <= 65532) global -> LDS ----
// D# per CDNA5 ISA ch.8: group0 {count=1|lds_addr|global_addr|type=2},
// group1 {data_size=1B, tensor_dim0=tile_dim0=bytes, stride=bytes}.
// This toolchain exposes the 6-arg builtin: (g0, g1, g2, g3, g4, cpol).
__device__ __forceinline__ void tdm_load_1d(unsigned lds_off,
                                            unsigned long long gaddr,
                                            unsigned bytes)
{
    v4u_ g0;
    g0.x = 1u;                                                   // count=1, user mode
    g0.y = lds_off;                                              // lds_addr (bytes)
    g0.z = (unsigned)(gaddr & 0xFFFFFFFFu);                      // global_addr[31:0]
    g0.w = (unsigned)((gaddr >> 32) & 0x1FFFFFFu) | (2u << 30);  // addr[56:32] | type=2
    v8i_ g1;
    g1[0] = 0;                                                   // wg_mask=0, data_size=1B
    g1[1] = (int)((bytes & 0xFFFFu) << 16);                      // tensor_dim0[15:0]
    g1[2] = (int)(((bytes >> 16) & 0xFFFFu) | (1u << 16));       // tensor_dim0[31:16], tdim1=1
    g1[3] = (int)((bytes & 0xFFFFu) << 16);                      // tile_dim0 = bytes
    g1[4] = 0;                                                   // tile_dim1/2 = 0 (1-D)
    g1[5] = (int)bytes;                                          // tensor_dim0_stride
    g1[6] = 0;
    g1[7] = 0;
    v4i_ z4 = {0, 0, 0, 0};
    v8i_ z8 = {0, 0, 0, 0, 0, 0, 0, 0};
    __builtin_amdgcn_tensor_load_to_lds(g0, g1, z4, z4, z8, 0);
}

// ---------------- kernel 3: fused top-p + feat + WMMA proto scoring ----------------
__global__ __launch_bounds__(256) void proto_main_kernel(
    const float* __restrict__ base_b, const float* __restrict__ S,
    const float* __restrict__ XY,     const float* __restrict__ P,
    const float* __restrict__ slot_mask, const float* __restrict__ alpha_param,
    const _Float16* __restrict__ ch,  const float* __restrict__ c2,
    const float* __restrict__ w,      float* __restrict__ out)
{
    __shared__ __align__(16) _Float16 featLds[M_ * D_];    // 16 KB features (f16)
    __shared__ __align__(16) _Float16 bBuf[2][32 * D_];    // 2 x 8 KB B double-buffer
    __shared__ float f2s[M_];
    __shared__ float sel[M_];
    __shared__ float sv[M_];
    __shared__ int   si[M_];
    __shared__ float scores[C_];
    __shared__ int   kshared;

    const int b    = blockIdx.x;
    const int t    = threadIdx.x;
    const int lane = t & 31;
    const int wave = t >> 5;

    // ---- Phase A: score, bitonic sort (descending), top-p selection ----
    if (t < M_) { sv[t] = P[b * M_ + t] * slot_mask[b * M_ + t]; si[t] = t; }
    __syncthreads();
    for (int ks = 2; ks <= M_; ks <<= 1) {
        for (int j = ks >> 1; j > 0; j >>= 1) {
            if (t < M_) {
                int ixj = t ^ j;
                if (ixj > t) {
                    bool up = ((t & ks) == 0);
                    float a = sv[t], bb = sv[ixj];
                    bool sw = up ? (a < bb) : (a > bb);
                    if (sw) {
                        sv[t] = bb; sv[ixj] = a;
                        int tmp = si[t]; si[t] = si[ixj]; si[ixj] = tmp;
                    }
                }
            }
            __syncthreads();
        }
    }
    if (t == 0) {
        float total = 0.f;
        for (int i = 0; i < M_; ++i) total += sv[i];
        float cs = 0.f; int k = 0;
        for (int i = 0; i < M_; ++i) { cs += sv[i]; if (cs <= TOPP_ * (total + 1e-8f)) ++k; }
        kshared = (k < 1) ? 1 : k;
    }
    if (t < M_) sel[t] = 0.f;
    __syncthreads();
    if (t < kshared) sel[si[t]] = 1.f;     // active slots in ORIGINAL index order

    // ---- Phase B: normalized features into LDS as f16; f2 per slot ----
    for (int i = 0; i < 8; ++i) {
        int m = wave * 8 + i;
        const float* Srow = S + ((size_t)b * M_ + m) * DSLOT;
        float ss = 0.f;
        for (int dl = lane; dl < DSLOT; dl += 32) { float v = Srow[dl]; ss += v * v; }
        for (int off = 16; off; off >>= 1) ss += __shfl_xor(ss, off, 32);
        float nS   = sqrtf(ss);
        float invS = 1.f / fmaxf(nS, 1e-12f);
        for (int dl = lane; dl < DSLOT; dl += 32)
            featLds[m * D_ + dl] = (_Float16)(Srow[dl] * invS);
        if (lane == 0) {
            float x = XY[((size_t)b * M_ + m) * 2 + 0];
            float y = XY[((size_t)b * M_ + m) * 2 + 1];
            float nxy = sqrtf(x * x + y * y);
            float inv2 = 0.5f / fmaxf(nxy, 1e-12f);
            featLds[m * D_ + 126] = (_Float16)(x * inv2);
            featLds[m * D_ + 127] = (_Float16)(y * inv2);
            float a1 = nS  / fmaxf(nS,  1e-12f);
            float a2 = nxy / fmaxf(nxy, 1e-12f);
            f2s[m] = a1 * a1 + 0.25f * a2 * a2;
        }
    }
    if (t < C_) scores[t] = 0.f;
    __syncthreads();

    // ---- Phase C: WMMA tiles; A pinned in VGPRs, B TDM-staged through LDS ----
    // exp(-B*(f2 + c2 - 2 dot)) = [sel*e^{-B f2}]_row * [w*e^{-B c2}]_col * e^{2B dot}
    // A 16x32 f16 layout: lane L row M=L%16; VGPR v: K=(v<4?0:16)+2*(v&3)+(L>=16?8:0)
    // B 32x16 f16 layout: lane L col N=L%16; VGPR v: K=(v<4?0:16)+2*(v&3)+(L>=16?8:0)
    const int off8  = (lane >= 16) ? 8 : 0;
    const int off16 = (lane >= 16) ? 16 : 0;
    const int rbase = lane & 15;
    const int st    = wave & 3;        // slot tile for this wave
    const int pcol  = wave >> 2;       // which 16-row half of the 32-row chunk
    const int arow  = st * 16 + rbase;

    union frag { v16h h; uint4 q[2]; };
    frag Af[4];
    #pragma unroll
    for (int cc = 0; cc < 4; ++cc) {
        const _Float16* ap = &featLds[arow * D_ + 32 * cc + off8];
        Af[cc].q[0] = *(const uint4*)ap;
        Af[cc].q[1] = *(const uint4*)(ap + 16);
    }
    float ef[8];                                    // row factor: sel * e^{-beta*f2}
    #pragma unroll
    for (int j = 0; j < 8; ++j) {
        int row = st * 16 + j + off8;
        ef[j] = sel[row] * __expf(-BETA_ * f2s[row]);
    }

    // prologue: DMA chunk 0 (class 0: 32 proto rows = 8 KB contiguous) into buf 0
    if (wave == 0)
        tdm_load_1d((unsigned)(size_t)(void*)&bBuf[0][0],
                    (unsigned long long)(size_t)ch, 32 * D_ * 2);

    for (int i = 0; i < NCH_; ++i) {               // chunk i == class i
        int buf = i & 1;
        if (wave == 0) __builtin_amdgcn_s_wait_tensorcnt(0);   // chunk i landed
        __syncthreads();                                       // publish buf / retire buf^1
        if (wave == 0 && i + 1 < NCH_)
            tdm_load_1d((unsigned)(size_t)(void*)&bBuf[buf ^ 1][0],
                        (unsigned long long)(size_t)(ch + (size_t)(i + 1) * 32 * D_),
                        32 * D_ * 2);

        int n = i * 32 + pcol * 16 + rbase;        // global proto row for this lane
        const _Float16* brow = &bBuf[buf][(pcol * 16 + rbase) * D_];

        // load all 4 B fragments first (one clause, graduated dscnt waits) ...
        frag Bf[4];
        #pragma unroll
        for (int cc = 0; cc < 4; ++cc) {
            const _Float16* bp = brow + 32 * cc + off8;
            Bf[cc].q[0] = *(const uint4*)bp;
            Bf[cc].q[1] = *(const uint4*)(bp + 16);
        }
        // ... then drain the WMMA chain
        v8f acc = {0.f, 0.f, 0.f, 0.f, 0.f, 0.f, 0.f, 0.f};
        #pragma unroll
        for (int cc = 0; cc < 4; ++cc)
            acc = __builtin_amdgcn_wmma_f32_16x16x32_f16(
                      false, Af[cc].h, false, Bf[cc].h, (short)0, acc, false, false);

        float wc = w[n] * __expf(-BETA_ * c2[n]);  // col factor, once per job
        float part = 0.f;
        #pragma unroll
        for (int j = 0; j < 8; ++j)
            part += ef[j] * __expf((2.0f * BETA_) * acc[j]);
        part *= wc;
        for (int offs = 16; offs; offs >>= 1) part += __shfl_xor(part, offs, 32);
        if (lane == 0) atomicAdd(&scores[i], part);
    }
    __syncthreads();

    // ---- Phase D: masked mean + alpha mix ----
    if (t < C_) {
        float alpha = 1.f / (1.f + __expf(-alpha_param[0]));
        float sc = scores[t] / (float)kshared;
        out[b * C_ + t] = alpha * base_b[b * C_ + t] + (1.f - alpha) * sc;
    }
}

extern "C" void kernel_launch(void* const* d_in, const int* in_sizes, int n_in,
                              void* d_out, int out_size, void* d_ws, size_t ws_size,
                              hipStream_t stream) {
    const float* base_b      = (const float*)d_in[0];   // (256,62)
    const float* S_slots     = (const float*)d_in[1];   // (256,64,126)
    const float* XY          = (const float*)d_in[2];   // (256,64,2)
    const float* P           = (const float*)d_in[3];   // (256,64)
    const float* slot_mask   = (const float*)d_in[4];   // (256,64)
    const float* centers     = (const float*)d_in[5];   // (62,32,128)
    const float* psi         = (const float*)d_in[6];   // (62,32)
    const float* alpha_param = (const float*)d_in[7];   // scalar
    float* out = (float*)d_out;                          // (256,62) f32

    char* ws = (char*)d_ws;
    _Float16* ch = (_Float16*)ws;                        // N_*D_*2 = 507904 B
    float* c2    = (float*)(ws + 507904);                // 1984 f32
    float* w     = (float*)(ws + 507904 + 8192);         // 1984 f32

    prep_centers_kernel<<<N_, 128, 0, stream>>>(centers, ch, c2);
    softmax_psi_kernel<<<C_, KP_, 0, stream>>>(psi, w);
    proto_main_kernel<<<B_, 256, 0, stream>>>(base_b, S_slots, XY, P, slot_mask,
                                              alpha_param, ch, c2, w, out);
}
